// PairBiasAttention_47519518163036
// MI455X (gfx1250) — matrix operand order repaired and checked by reading, not compiled, not run on hardware
//
#include <hip/hip_runtime.h>
#include <stdint.h>

// ---------------------------------------------------------------------------
// PairBiasAttention for MI455X (gfx1250), wave32 + WMMA bf16 path.
// B=1, N=1536, D=512, P=64, H=16, DH=64, INNER=1024.
// HBM-bound on pair_feats (604 MB): stream it exactly once via the Tensor
// Data Mover into LDS, keep all GEMM operands bf16, accumulate f32 in
// v_wmma_f32_16x16x32_bf16.
// ---------------------------------------------------------------------------

namespace {
constexpr int N_     = 1536;
constexpr int D_     = 512;
constexpr int P_     = 64;
constexpr int H_     = 16;
constexpr int DH_    = 64;
constexpr int INNER_ = 1024;
}

typedef __bf16 bf16;
typedef __attribute__((ext_vector_type(16))) __bf16 v16bf;
typedef __attribute__((ext_vector_type(8)))  float  v8f;
typedef uint32_t u32x4 __attribute__((ext_vector_type(4)));
typedef uint32_t u32x8 __attribute__((ext_vector_type(8)));

static __device__ __forceinline__ v8f wmma_bf16(v16bf a, v16bf b, v8f c) {
  // (neg_a, A, neg_b, B, c_mod, C, reuse_a, reuse_b)
  return __builtin_amdgcn_wmma_f32_16x16x32_bf16(false, a, false, b, (short)0, c,
                                                 false, false);
}

// A fragment (16x32 bf16, M x K): lane holds row M=lane&15, 16 contiguous
// K values starting at base = (lane&16)?16:0 (matches B fragment order).
static __device__ __forceinline__ v16bf load_a_frag(const bf16* row, int lane) {
  const int base = (lane & 16) ? 16 : 0;
  v16bf f;
#pragma unroll
  for (int i = 0; i < 8; ++i) f[i] = row[base + i];
#pragma unroll
  for (int i = 0; i < 8; ++i) f[i + 8] = row[base + 8 + i];
  return f;
}

// B fragment (32x16 bf16, K x N): lane holds column N=lane&15, 16 contiguous
// K values starting at (lane&16)?16:0 (same k order as the A fragment).
static __device__ __forceinline__ v16bf load_b_frag(const bf16* col, int lane) {
  const int base = (lane & 16) ? 16 : 0;
  v16bf f;
#pragma unroll
  for (int i = 0; i < 16; ++i) f[i] = col[base + i];
  return f;
}

// ---------------------------------------------------------------------------
// LN over node feats (D=512) -> bf16 [N,D]
// ---------------------------------------------------------------------------
__global__ void ln_node_kernel(const float* __restrict__ x,
                               const float* __restrict__ g,
                               const float* __restrict__ b,
                               bf16* __restrict__ out) {
  __shared__ float red[256];
  const int row = blockIdx.x;
  const int tid = threadIdx.x;
  const float* xr = x + (size_t)row * D_;
  float v0 = xr[tid], v1 = xr[tid + 256];
  red[tid] = v0 + v1; __syncthreads();
  for (int o = 128; o > 0; o >>= 1) { if (tid < o) red[tid] += red[tid + o]; __syncthreads(); }
  float mean = red[0] * (1.0f / D_);
  __syncthreads();
  float d0 = v0 - mean, d1 = v1 - mean;
  red[tid] = d0 * d0 + d1 * d1; __syncthreads();
  for (int o = 128; o > 0; o >>= 1) { if (tid < o) red[tid] += red[tid + o]; __syncthreads(); }
  float rstd = rsqrtf(red[0] * (1.0f / D_) + 1e-5f);
  bf16* orow = out + (size_t)row * D_;
  orow[tid]       = (bf16)(d0 * rstd * g[tid] + b[tid]);
  orow[tid + 256] = (bf16)(d1 * rstd * g[tid + 256] + b[tid + 256]);
}

// ---------------------------------------------------------------------------
// W [K,Ncols] f32 -> Wt [Ncols,K] bf16 (so B-fragments load contiguously)
// ---------------------------------------------------------------------------
__global__ void convert_w_kernel(const float* __restrict__ W, bf16* __restrict__ Wt,
                                 int K, int Ncols) {
  int idx = blockIdx.x * 256 + threadIdx.x;
  if (idx >= K * Ncols) return;
  int k = idx / Ncols, n = idx % Ncols;
  Wt[(size_t)n * K + k] = (bf16)W[idx];
}

// ---------------------------------------------------------------------------
// C[M,Ncols] f32 = A[M,K] bf16 @ Bt[Ncols,K]^T + bias.  One 16x16 tile/wave,
// 8 waves/block, K stepped by 32 (one WMMA per step).
// ---------------------------------------------------------------------------
__global__ void gemm_bf16_kernel(const bf16* __restrict__ A,
                                 const bf16* __restrict__ Bt,
                                 const float* __restrict__ bias,
                                 float* __restrict__ C,
                                 int M, int K, int Ncols) {
  const int lane = threadIdx.x & 31;
  const int wave = threadIdx.x >> 5;
  const int tile = blockIdx.x * 8 + wave;
  const int tilesN = Ncols >> 4;
  const int tm = (tile / tilesN) * 16;
  const int tn = (tile % tilesN) * 16;
  if (tm >= M) return;
  const int r = lane & 15;
  const bf16* arow = A + (size_t)(tm + r) * K;
  const bf16* brow = Bt + (size_t)(tn + r) * K;
  v8f acc = {};
  for (int k0 = 0; k0 < K; k0 += 32)
    acc = wmma_bf16(load_a_frag(arow + k0, lane), load_b_frag(brow + k0, lane), acc);
  const int mrow = tm + ((lane & 16) ? 8 : 0);
  const float bn = bias ? bias[tn + r] : 0.0f;
#pragma unroll
  for (int j = 0; j < 8; ++j)
    C[(size_t)(mrow + j) * Ncols + tn + r] = acc[j] + bn;
}

// ---------------------------------------------------------------------------
// LN(q_pre), LN(k), split heads: q,k -> bf16 [H][N][DH]; v -> bf16 [H][DH][N]
// ---------------------------------------------------------------------------
__global__ void qkv_split_kernel(const float* __restrict__ qpre,
                                 const float* __restrict__ kv,
                                 const float* __restrict__ qg, const float* __restrict__ qb,
                                 const float* __restrict__ kg, const float* __restrict__ kb,
                                 bf16* __restrict__ qbf, bf16* __restrict__ kbf,
                                 bf16* __restrict__ vt) {
  __shared__ float red[256];
  const int row = blockIdx.x, tid = threadIdx.x;
  const float* qr = qpre + (size_t)row * INNER_;
  const float* kr = kv + (size_t)row * (2 * INNER_);
  const float* vr = kr + INNER_;
  float qv[4], kvv[4];
#pragma unroll
  for (int i = 0; i < 4; ++i) { qv[i] = qr[tid * 4 + i]; kvv[i] = kr[tid * 4 + i]; }

  auto blk_sum = [&](float v) -> float {
    red[tid] = v; __syncthreads();
    for (int o = 128; o > 0; o >>= 1) { if (tid < o) red[tid] += red[tid + o]; __syncthreads(); }
    float r = red[0]; __syncthreads(); return r;
  };

  float qmean = blk_sum(qv[0] + qv[1] + qv[2] + qv[3]) * (1.0f / INNER_);
  float kmean = blk_sum(kvv[0] + kvv[1] + kvv[2] + kvv[3]) * (1.0f / INNER_);
  float qd[4], kd[4], qss = 0.f, kss = 0.f;
#pragma unroll
  for (int i = 0; i < 4; ++i) {
    qd[i] = qv[i] - qmean; qss += qd[i] * qd[i];
    kd[i] = kvv[i] - kmean; kss += kd[i] * kd[i];
  }
  float qrstd = rsqrtf(blk_sum(qss) * (1.0f / INNER_) + 1e-5f);
  float krstd = rsqrtf(blk_sum(kss) * (1.0f / INNER_) + 1e-5f);
#pragma unroll
  for (int i = 0; i < 4; ++i) {
    int c = tid * 4 + i;
    int h = c >> 6, dh = c & 63;
    qbf[((size_t)h * N_ + row) * DH_ + dh] = (bf16)(qd[i] * qrstd * qg[c] + qb[c]);
    kbf[((size_t)h * N_ + row) * DH_ + dh] = (bf16)(kd[i] * krstd * kg[c] + kb[c]);
    vt[((size_t)(h * DH_ + dh)) * N_ + row] = (bf16)vr[c];
  }
}

// ---------------------------------------------------------------------------
// bias[h][i][j] bf16 = LN(pair[i,j,:]) @ Wbias[:,h], mask folded as -1e30.
// Per block: TDM (tensor_load_to_lds) stages a [128,64] f32 pair tile into
// LDS (TENSORcnt-tracked), then 8 waves each LN 16 rows and WMMA vs Wbias.
// ---------------------------------------------------------------------------
__global__ void pair_bias_kernel(const float* __restrict__ pair,
                                 const uint8_t* __restrict__ mask,
                                 const float* __restrict__ pg, const float* __restrict__ pb,
                                 const float* __restrict__ Wbias,
                                 bf16* __restrict__ biasb) {
  __shared__ float plds[128][64];   // 32 KB tile, written by the TDM
  __shared__ bf16 pbuf[8][16][72];  // 72: pad rows to stagger banks
  const int lane = threadIdx.x & 31;
  const int wave = threadIdx.x >> 5;
  const int ij0 = blockIdx.x * 128;          // 128 flattened (i,j) rows / block
  const int r = lane & 15;
  const int hi = (lane >> 4) & 1;

  // ---- TDM: one wave issues the tensor DMA for the whole block tile -------
  if (wave == 0) {
    const uint64_t ga = (uint64_t)(uintptr_t)(pair + (size_t)ij0 * P_);
    // LDS flat addresses carry the LDS byte offset in the low 32 bits.
    const uint32_t lds_addr = (uint32_t)(uintptr_t)(&plds[0][0]);
    u32x4 g0;
    g0[0] = 1u;                                   // count=1, user descriptor
    g0[1] = lds_addr;                             // lds_addr [63:32]
    g0[2] = (uint32_t)ga;                         // global_addr [95:64]
    g0[3] = (uint32_t)(ga >> 32) | (2u << 30);    // addr hi + type=2 ("image")
    u32x8 g1;
    g1[0] = (2u << 16);        // workgroup_mask=0, data_size=2 (4 bytes)
    g1[1] = ((uint32_t)P_ << 16);                 // tensor_dim0 = 64 (lo16)
    g1[2] = 0u;                // tensor_dim0 hi, tensor_dim1 lo16 (N*N lo16=0)
    g1[3] = (uint32_t)(((uint64_t)N_ * N_) >> 16) // tensor_dim1 hi16
            | ((uint32_t)P_ << 16);               // tile_dim0 = 64
    g1[4] = 128u;              // tile_dim1 = 128 rows, tile_dim2 = 0
    g1[5] = (uint32_t)P_;      // tensor_dim0_stride = 64 (lo32)
    g1[6] = 0u;                // stride hi, tensor_dim1_stride lo
    g1[7] = 0u;
    asm volatile("tensor_load_to_lds %0, %1" :: "s"(g0), "s"(g1) : "memory");
    __builtin_amdgcn_s_wait_tensorcnt(0);
  }
  __syncthreads();

  // ---- LN over P=64 from LDS: lane handles row r, half hi (32 elems) ------
  const float* prow = &plds[wave * 16 + r][hi * 32];
  float vals[32];
  float s = 0.f;
#pragma unroll
  for (int i = 0; i < 32; ++i) { vals[i] = prow[i]; s += vals[i]; }
  s += __shfl_xor(s, 16, 32);
  float mean = s * (1.0f / 64.0f);
  float ss = 0.f;
#pragma unroll
  for (int i = 0; i < 32; ++i) { float d = vals[i] - mean; ss += d * d; }
  ss += __shfl_xor(ss, 16, 32);
  float rstd = rsqrtf(ss * (1.0f / 64.0f) + 1e-5f);
#pragma unroll
  for (int i = 0; i < 32; ++i) {
    int c = hi * 32 + i;
    pbuf[wave][r][c] = (bf16)((vals[i] - mean) * rstd * pg[c] + pb[c]);
  }

  // Wbias B-fragments (cols = heads), K = P = 64 -> two fragments.
  v16bf wb0, wb1;
  {
    const int base = (lane & 16) ? 16 : 0;
#pragma unroll
    for (int i = 0; i < 16; ++i) {
      wb0[i] = (bf16)Wbias[(base + i) * H_ + r];
      wb1[i] = (bf16)Wbias[(32 + base + i) * H_ + r];
    }
  }
  v16bf a0 = load_a_frag(&pbuf[wave][r][0], lane);
  v16bf a1 = load_a_frag(&pbuf[wave][r][32], lane);
  v8f acc = {};
  acc = wmma_bf16(a0, wb0, acc);
  acc = wmma_bf16(a1, wb1, acc);

  // D layout: lane col n = head; rows j+8*hi = flattened ij within the tile.
#pragma unroll
  for (int j = 0; j < 8; ++j) {
    int ij = ij0 + wave * 16 + j + 8 * hi;
    float v = mask[ij] ? acc[j] : -1e30f;
    biasb[(size_t)r * N_ * N_ + ij] = (bf16)v;
  }
}

// ---------------------------------------------------------------------------
// Fused flash attention: 1 wave per (head, 16-query tile), 32 keys/iter.
// sim (4 WMMA, K=DH=64) -> +bias -> online softmax (shfl row-reduce in the
// C/D layout) -> P via LDS re-layout -> PV (4 WMMA) -> sigmoid(g) gate.
// Next-iteration K/V tiles prefetched with global_prefetch_b8.
// ---------------------------------------------------------------------------
__global__ void flash_attn_kernel(const bf16* __restrict__ qbf,
                                  const bf16* __restrict__ kbf,
                                  const bf16* __restrict__ vt,
                                  const bf16* __restrict__ biasb,
                                  const float* __restrict__ g,
                                  bf16* __restrict__ obf) {
  __shared__ bf16 pl[4][16][40];  // per-wave 16x32 P tile, padded
  const int lane = threadIdx.x & 31;
  const int wave = threadIdx.x >> 5;
  const int job = blockIdx.x * 4 + wave;
  const int h = job / (N_ / 16);
  const int qt = job % (N_ / 16);
  const int n = lane & 15;
  const int hi = (lane >> 4) & 1;

  const bf16* qrow = qbf + ((size_t)h * N_ + qt * 16 + n) * DH_;
  const v16bf qa0 = load_a_frag(qrow, lane);        // dh 0..31
  const v16bf qa1 = load_a_frag(qrow + 32, lane);   // dh 32..63

  v8f o0 = {}, o1 = {}, o2 = {}, o3 = {};
  float m8[8], l8[8];
#pragma unroll
  for (int j = 0; j < 8; ++j) { m8[j] = -1e30f; l8[j] = 0.f; }

  const bf16* kh  = kbf + (size_t)h * N_ * DH_;
  const bf16* vth = vt + (size_t)h * DH_ * N_;
  const bf16* bh  = biasb + (size_t)h * N_ * N_;
  const float scale = 0.125f;  // DH^-0.5

  for (int jb = 0; jb < N_; jb += 32) {
    // Prefetch next iteration's K/V cachelines (global_prefetch_b8).
    if (jb + 32 < N_) {
      __builtin_prefetch(kh + (size_t)(jb + 32 + n) * DH_, 0, 0);
      __builtin_prefetch(kh + (size_t)(jb + 48 + n) * DH_, 0, 0);
      __builtin_prefetch(vth + (size_t)(n) * N_ + jb + 32, 0, 0);
      __builtin_prefetch(vth + (size_t)(32 + n) * N_ + jb + 32, 0, 0);
    }

    // S = Q K^T for two 16-key tiles (B cols = keys, B k-dim = dh)
    const bf16* kc0 = kh + (size_t)(jb + n) * DH_;
    const bf16* kc1 = kh + (size_t)(jb + 16 + n) * DH_;
    v8f s0 = {}, s1 = {};
    s0 = wmma_bf16(qa0, load_b_frag(kc0, lane), s0);
    s0 = wmma_bf16(qa1, load_b_frag(kc0 + 32, lane), s0);
    s1 = wmma_bf16(qa0, load_b_frag(kc1, lane), s1);
    s1 = wmma_bf16(qa1, load_b_frag(kc1 + 32, lane), s1);

    const size_t brow = (size_t)(qt * 16 + hi * 8) * N_ + jb + n;
#pragma unroll
    for (int j = 0; j < 8; ++j) {
      s0[j] = s0[j] * scale + (float)bh[brow + (size_t)j * N_];
      s1[j] = s1[j] * scale + (float)bh[brow + (size_t)j * N_ + 16];
    }

    // Online softmax. Row m'=j+8*hi lives across the 16 lanes of one half.
    float p0[8], p1[8];
#pragma unroll
    for (int j = 0; j < 8; ++j) {
      float t = fmaxf(s0[j], s1[j]);
      t = fmaxf(t, __shfl_xor(t, 1, 32));
      t = fmaxf(t, __shfl_xor(t, 2, 32));
      t = fmaxf(t, __shfl_xor(t, 4, 32));
      t = fmaxf(t, __shfl_xor(t, 8, 32));
      float mn = fmaxf(m8[j], t);
      float corr = __expf(m8[j] - mn);
      float e0 = __expf(s0[j] - mn);
      float e1 = __expf(s1[j] - mn);
      float rs = e0 + e1;
      rs += __shfl_xor(rs, 1, 32);
      rs += __shfl_xor(rs, 2, 32);
      rs += __shfl_xor(rs, 4, 32);
      rs += __shfl_xor(rs, 8, 32);
      l8[j] = l8[j] * corr + rs;
      m8[j] = mn;
      o0[j] *= corr; o1[j] *= corr; o2[j] *= corr; o3[j] *= corr;
      p0[j] = e0; p1[j] = e1;
    }

    // Re-layout P (C/D layout -> A layout) through LDS.
#pragma unroll
    for (int j = 0; j < 8; ++j) {
      pl[wave][j + 8 * hi][n]      = (bf16)p0[j];
      pl[wave][j + 8 * hi][16 + n] = (bf16)p1[j];
    }
    v16bf pfrag = load_a_frag(&pl[wave][n][0], lane);

    // O += P @ V (V cols = dh chunks, V k-dim = keys; vt rows contiguous in j)
    o0 = wmma_bf16(pfrag, load_b_frag(vth + (size_t)(0 * 16 + n) * N_ + jb, lane), o0);
    o1 = wmma_bf16(pfrag, load_b_frag(vth + (size_t)(1 * 16 + n) * N_ + jb, lane), o1);
    o2 = wmma_bf16(pfrag, load_b_frag(vth + (size_t)(2 * 16 + n) * N_ + jb, lane), o2);
    o3 = wmma_bf16(pfrag, load_b_frag(vth + (size_t)(3 * 16 + n) * N_ + jb, lane), o3);
  }

  // Normalize, gate with sigmoid(g), store bf16 o [N][INNER].
  const int qbase = qt * 16 + 8 * hi;
#pragma unroll
  for (int j = 0; j < 8; ++j) {
    const int row = qbase + j;
    const float inv = 1.0f / (l8[j] + 1e-20f);
    const size_t gb = (size_t)row * INNER_ + h * DH_ + n;
    float g0 = g[gb + 0], g1 = g[gb + 16], g2 = g[gb + 32], g3 = g[gb + 48];
    obf[gb + 0]  = (bf16)(o0[j] * inv * (1.f / (1.f + __expf(-g0))));
    obf[gb + 16] = (bf16)(o1[j] * inv * (1.f / (1.f + __expf(-g1))));
    obf[gb + 32] = (bf16)(o2[j] * inv * (1.f / (1.f + __expf(-g2))));
    obf[gb + 48] = (bf16)(o3[j] * inv * (1.f / (1.f + __expf(-g3))));
  }
}

// ---------------------------------------------------------------------------
extern "C" void kernel_launch(void* const* d_in, const int* in_sizes, int n_in,
                              void* d_out, int out_size, void* d_ws, size_t ws_size,
                              hipStream_t stream) {
  (void)in_sizes; (void)n_in; (void)out_size; (void)ws_size;
  const float*   node  = (const float*)d_in[0];
  const float*   pair  = (const float*)d_in[1];
  const uint8_t* mask  = (const uint8_t*)d_in[2];
  const float*   ln_ng = (const float*)d_in[3];
  const float*   ln_nb = (const float*)d_in[4];
  const float*   ln_pg = (const float*)d_in[5];
  const float*   ln_pb = (const float*)d_in[6];
  const float*   Wq    = (const float*)d_in[7];
  const float*   bq    = (const float*)d_in[8];
  const float*   Wkv   = (const float*)d_in[9];
  const float*   bkv   = (const float*)d_in[10];
  const float*   qg    = (const float*)d_in[11];
  const float*   qb    = (const float*)d_in[12];
  const float*   kg    = (const float*)d_in[13];
  const float*   kb    = (const float*)d_in[14];
  const float*   Wg    = (const float*)d_in[15];
  const float*   bg    = (const float*)d_in[16];
  const float*   Wbias = (const float*)d_in[17];
  const float*   Wout  = (const float*)d_in[18];
  const float*   bout  = (const float*)d_in[19];

  char* ws = (char*)d_ws;
  size_t off = 0;
  auto alloc = [&](size_t bytes) -> void* {
    void* p = ws + off;
    off = (off + bytes + 255) & ~(size_t)255;
    return p;
  };
  bf16*  xbf   = (bf16*)alloc((size_t)N_ * D_ * 2);
  bf16*  wqT   = (bf16*)alloc((size_t)INNER_ * D_ * 2);
  bf16*  wkvT  = (bf16*)alloc((size_t)2 * INNER_ * D_ * 2);
  bf16*  wgT   = (bf16*)alloc((size_t)INNER_ * D_ * 2);
  bf16*  woT   = (bf16*)alloc((size_t)D_ * INNER_ * 2);
  float* qpre  = (float*)alloc((size_t)N_ * INNER_ * 4);
  float* kvb   = (float*)alloc((size_t)N_ * 2 * INNER_ * 4);
  float* gbuf  = (float*)alloc((size_t)N_ * INNER_ * 4);
  bf16*  qbf   = (bf16*)alloc((size_t)H_ * N_ * DH_ * 2);
  bf16*  kbf   = (bf16*)alloc((size_t)H_ * N_ * DH_ * 2);
  bf16*  vtb   = (bf16*)alloc((size_t)H_ * DH_ * N_ * 2);
  bf16*  obf   = (bf16*)alloc((size_t)N_ * INNER_ * 2);
  bf16*  biasb = (bf16*)alloc((size_t)H_ * N_ * N_ * 2);  // 75.5 MB

  ln_node_kernel<<<N_, 256, 0, stream>>>(node, ln_ng, ln_nb, xbf);
  convert_w_kernel<<<(D_ * INNER_ + 255) / 256, 256, 0, stream>>>(Wq, wqT, D_, INNER_);
  convert_w_kernel<<<(D_ * 2 * INNER_ + 255) / 256, 256, 0, stream>>>(Wkv, wkvT, D_, 2 * INNER_);
  convert_w_kernel<<<(D_ * INNER_ + 255) / 256, 256, 0, stream>>>(Wg, wgT, D_, INNER_);
  convert_w_kernel<<<(INNER_ * D_ + 255) / 256, 256, 0, stream>>>(Wout, woT, INNER_, D_);

  gemm_bf16_kernel<<<(N_ / 16) * (INNER_ / 16) / 8, 256, 0, stream>>>(
      xbf, wqT, bq, qpre, N_, D_, INNER_);
  gemm_bf16_kernel<<<(N_ / 16) * (2 * INNER_ / 16) / 8, 256, 0, stream>>>(
      xbf, wkvT, bkv, kvb, N_, D_, 2 * INNER_);
  gemm_bf16_kernel<<<(N_ / 16) * (INNER_ / 16) / 8, 256, 0, stream>>>(
      xbf, wgT, bg, gbuf, N_, D_, INNER_);

  qkv_split_kernel<<<N_, 256, 0, stream>>>(qpre, kvb, qg, qb, kg, kb, qbf, kbf, vtb);

  pair_bias_kernel<<<N_ * N_ / 128, 256, 0, stream>>>(
      pair, mask, ln_pg, ln_pb, Wbias, biasb);

  flash_attn_kernel<<<(H_ * (N_ / 16)) / 4, 128, 0, stream>>>(
      qbf, kbf, vtb, biasb, gbuf, obf);

  gemm_bf16_kernel<<<(N_ / 16) * (D_ / 16) / 8, 256, 0, stream>>>(
      obf, woT, bout, (float*)d_out, N_, INNER_, D_);
}
